// GroupedQueryAttention_87479893885354
// MI455X (gfx1250) — matrix operand order repaired and checked by reading, not compiled, not run on hardware
//
#include <hip/hip_runtime.h>

// ---------------------------------------------------------------------------
// GQA attention for MI455X (gfx1250, wave32, WMMA 16x16x32 bf16 + TDM)
// B=1, S=4096, D=1024, H=16, DK=64, G=4, HKV=4, DKV=256
// ---------------------------------------------------------------------------

#define S_LEN 4096
#define D_DIM 1024
#define H_NUM 16
#define DK_DIM 64
#define DKV_DIM 256
#define ATTN_SCALE 0.125f  // 1/sqrt(64)
#define NEG_BIG (-1.0e30f)

typedef __attribute__((ext_vector_type(16))) __bf16 v16bf;
typedef __attribute__((ext_vector_type(8)))  float  v8f;
typedef __attribute__((ext_vector_type(4)))  unsigned int uv4;   // 128-bit move
typedef __attribute__((ext_vector_type(8)))  int iv8;
typedef __attribute__((ext_vector_type(4)))  int iv4;

typedef __attribute__((address_space(3))) __bf16 lbf16;
typedef __attribute__((address_space(3))) const uv4 lds_cuv4;

static __device__ __forceinline__ __bf16 f2bf(float f) {
  unsigned int u = __builtin_bit_cast(unsigned int, f);
  u += 0x7FFFu + ((u >> 16) & 1u);  // round-to-nearest-even
  unsigned short s = (unsigned short)(u >> 16);
  return __builtin_bit_cast(__bf16, s);
}

// A-operand fragment (16x32 bf16), contiguous k order (same as B fragments):
// lane m=lane&15 holds row m; element i -> k = i + 16*(lane>=16)
static __device__ __forceinline__ v16bf load_frag_a(const __bf16* __restrict__ base,
                                                    int ld, int row0, int k0) {
  const int lane = threadIdx.x & 31;
  const int m = lane & 15;
  const int hi = lane >> 4;
  const __bf16* p = base + (size_t)(row0 + m) * ld + k0 + hi * 16;
  union { v16bf v; uv4 u[2]; } r;
  r.u[0] = *(const uv4*)(p);
  r.u[1] = *(const uv4*)(p + 8);
  return r.v;
}

// B-operand fragment (32x16 bf16) from [N,K] row-major (B^T) in global memory.
// contiguous k order (same as A fragments): lane n=lane&15 holds column n; element i -> k = i + 16*(lane>=16)
static __device__ __forceinline__ v16bf load_frag_b(const __bf16* __restrict__ baseT,
                                                    int ld, int n0, int k0) {
  const int lane = threadIdx.x & 31;
  const int n = lane & 15;
  const int hi = lane >> 4;
  const __bf16* p = baseT + (size_t)(n0 + n) * ld + k0 + hi * 16;
  union { v16bf v; uv4 u[2]; } r;
  r.u[0] = *(const uv4*)(p);
  r.u[1] = *(const uv4*)(p + 8);
  return r.v;
}

// Same B fragment but from an LDS-staged tile (rows at 40-bf16 / 80B stride:
// 16B-aligned vector loads, conflict-free across 16 rows). ds_load_b128 path.
#define BT_STRIDE 40
static __device__ __forceinline__ v16bf load_frag_b_lds(unsigned lds_off, int n0) {
  const int lane = threadIdx.x & 31;
  const int n = lane & 15;
  const int hi = lane >> 4;
  const lbf16* p = (const lbf16*)(size_t)lds_off + (size_t)(n0 + n) * BT_STRIDE + hi * 16;
  union { v16bf v; uv4 u[2]; } r;
  r.u[0] = *(lds_cuv4*)(p);
  r.u[1] = *(lds_cuv4*)(p + 8);
  return r.v;
}

static __device__ __forceinline__ v8f wmma_bf16(v16bf a, v16bf b, v8f c) {
  return __builtin_amdgcn_wmma_f32_16x16x32_bf16(false, a, false, b, (short)0, c,
                                                 false, false);
}

// ---------------------- Tensor Data Mover (gfx1250) ------------------------
#if __has_builtin(__builtin_amdgcn_tensor_load_to_lds)
#define HAVE_TDM 1
// DMA one B^T tile [64 rows x 32 K] of bf16 into LDS, padding 16B per 64B row
// so LDS rows land at 80B stride. D# per CDNA5 ISA ch.8.
static __device__ __forceinline__ void tdm_load_tile_b(const __bf16* src_tile,
                                                       unsigned lds_off,
                                                       unsigned K_dim,
                                                       unsigned N_dim) {
  const unsigned long long ga = (unsigned long long)(const void*)src_tile;
  uv4 g0;
  g0[0] = 1u;                                            // count=1 (valid D#)
  g0[1] = lds_off;                                       // lds_addr
  g0[2] = (unsigned)(ga & 0xFFFFFFFFu);                  // global_addr[31:0]
  g0[3] = (unsigned)((ga >> 32) & 0x1FFFFFFu) | (2u << 30);  // addr[56:32], type=2

  iv8 g1;
  g1[0] = (int)((1u << 16)      // data_size = 2 bytes (bf16)
              | (1u << 20)      // pad_enable
              | (3u << 22)      // pad_interval: every 16 DWORDs (64B)
              | (3u << 25));    // pad_amount: 4 DWORDs (16B)
  g1[1] = (int)((K_dim & 0xFFFFu) << 16);                      // tensor_dim0 lo
  g1[2] = (int)(((K_dim >> 16) & 0xFFFFu) | ((N_dim & 0xFFFFu) << 16));
  g1[3] = (int)(((N_dim >> 16) & 0xFFFFu) | (32u << 16));      // tile_dim0 = 32
  g1[4] = 64;                                                  // tile_dim1 = 64
  g1[5] = (int)K_dim;                                          // dim0_stride lo
  g1[6] = 0;
  g1[7] = 0;
  const iv4 z4 = {0, 0, 0, 0};
#if defined(__clang_major__) && __clang_major__ >= 23
  const iv8 z8 = {0, 0, 0, 0, 0, 0, 0, 0};
  __builtin_amdgcn_tensor_load_to_lds(g0, g1, z4, z4, z8, 0);
#else
  __builtin_amdgcn_tensor_load_to_lds(g0, g1, z4, z4, 0);
#endif
}
#else
#define HAVE_TDM 0
#endif

// --------------------------- prep kernels ----------------------------------

__global__ void convert_f32_to_bf16_kernel(const float* __restrict__ in,
                                           __bf16* __restrict__ out, int n) {
  for (int i = blockIdx.x * blockDim.x + threadIdx.x; i < n;
       i += gridDim.x * blockDim.x)
    out[i] = f2bf(in[i]);
}

// in: [R][C] f32 row-major -> out: [C][R] bf16 row-major (R,C multiples of 32)
__global__ void transpose_to_bf16_kernel(const float* __restrict__ in,
                                         __bf16* __restrict__ out, int R, int C) {
  __shared__ float tile[32][33];
  const int rb = blockIdx.y * 32, cb = blockIdx.x * 32;
  const int tx = threadIdx.x, ty = threadIdx.y;  // 32 x 8
  for (int i = ty; i < 32; i += 8)
    tile[i][tx] = in[(size_t)(rb + i) * C + cb + tx];
  __syncthreads();
  for (int i = ty; i < 32; i += 8)
    out[(size_t)(cb + i) * R + rb + tx] = f2bf(tile[tx][i]);
}

// ------------------------------ GEMM ---------------------------------------
// C[M,N] = A[M,K](bf16) * B[K,N] + bias, B supplied transposed Bt[N,K].
// Block = 256 threads (8 waves); block tile 128x64. The 64x32 B^T tile is
// staged in LDS once per block (TDM async DMA, double-buffered, synchronized
// with s_wait_tensorcnt + barrier); A fragments are register double-buffered.
// mode 0: bf16 row-major; mode 1: bf16 transposed [N,M]; mode 2: f32 row-major.
__global__ __launch_bounds__(256) void gemm_bf16_kernel(
    const __bf16* __restrict__ A, const __bf16* __restrict__ Bt,
    const float* __restrict__ bias, int M, int N, int K,
    void* __restrict__ outp, int mode) {
  __shared__ __align__(16) __bf16 btile[2][64 * BT_STRIDE];

  const int wave = threadIdx.x >> 5;
  const int lane = threadIdx.x & 31;
  const int hi = lane >> 4, n16 = lane & 15;
  const int row0 = blockIdx.y * 128 + wave * 16;
  const int col0 = blockIdx.x * 64;
  const int nk = K >> 5;

  unsigned loff[2];
  loff[0] = (unsigned)(size_t)(lbf16*)&btile[0][0];
  loff[1] = (unsigned)(size_t)(lbf16*)&btile[1][0];
  // launder so the never-stored (DMA-written) LDS can't be folded away
  asm volatile("" : "+v"(loff[0]), "+v"(loff[1]) :: "memory");

  v8f acc[4] = {v8f{}, v8f{}, v8f{}, v8f{}};

#if HAVE_TDM
  if (wave == 0)
    tdm_load_tile_b(Bt + (size_t)col0 * K, loff[0], (unsigned)K, (unsigned)N);
#endif
  v16bf a_cur = load_frag_a(A, K, row0, 0);

  for (int i = 0; i < nk; ++i) {
    const int k0 = i << 5;
#if HAVE_TDM
    __builtin_amdgcn_s_wait_tensorcnt(0);   // tile i resident in LDS
    __syncthreads();                        // visible to all waves; buf free
    if (wave == 0 && i + 1 < nk)
      tdm_load_tile_b(Bt + (size_t)col0 * K + (k0 + 32), loff[(i + 1) & 1],
                      (unsigned)K, (unsigned)N);
#else
    // fallback: cooperative copy of the 64x32 tile into padded LDS
    {
      const int t = threadIdx.x;           // 256 threads, one 16B chunk each
      const int r = t >> 2, c = (t & 3) * 8;
      *(uv4*)(&btile[i & 1][r * BT_STRIDE + c]) =
          *(const uv4*)(Bt + (size_t)(col0 + r) * K + k0 + c);
    }
    __syncthreads();
#endif

    if (k0 + 64 <= K)
      __builtin_prefetch(A + (size_t)(row0 + n16) * K + k0 + 64, 0, 1);

    v16bf a_nxt = a_cur;
    if (i + 1 < nk) a_nxt = load_frag_a(A, K, row0, k0 + 32);

    // batch all 4 B-fragment LDS loads, then the 4 WMMAs (one dscnt wait)
    const unsigned cur = loff[i & 1];
    v16bf b0 = load_frag_b_lds(cur, 0);
    v16bf b1 = load_frag_b_lds(cur, 16);
    v16bf b2 = load_frag_b_lds(cur, 32);
    v16bf b3 = load_frag_b_lds(cur, 48);
    acc[0] = wmma_bf16(a_cur, b0, acc[0]);
    acc[1] = wmma_bf16(a_cur, b1, acc[1]);
    acc[2] = wmma_bf16(a_cur, b2, acc[2]);
    acc[3] = wmma_bf16(a_cur, b3, acc[3]);
    a_cur = a_nxt;
#if !HAVE_TDM
    __syncthreads();
#endif
  }

#pragma unroll
  for (int j = 0; j < 4; ++j) {
    const int col = col0 + j * 16 + n16;
    const float bv = bias ? bias[col] : 0.0f;
#pragma unroll
    for (int v = 0; v < 8; ++v) {
      const int row = row0 + v + 8 * hi;
      const float val = acc[j][v] + bv;
      if (mode == 2)
        ((float*)outp)[(size_t)row * N + col] = val;
      else if (mode == 1)
        ((__bf16*)outp)[(size_t)col * M + row] = f2bf(val);
      else
        ((__bf16*)outp)[(size_t)row * N + col] = f2bf(val);
    }
  }
}

// --------------------------- attention -------------------------------------
// Flash-style online-softmax GQA. Grid: (S/128, H). 8 waves/block; each wave
// owns a 16-row query strip of one head and streams keys in 32-wide tiles.
// K fragments are register double-buffered across tiles and V fragments are
// issued before the softmax VALU stretch so global latency overlaps ALU work.
#define P_STRIDE 40  // bf16 elems per LDS row: 80B, 16B-aligned, bank-clean

__global__ __launch_bounds__(256) void gqa_attn_kernel(
    const __bf16* __restrict__ Qb,          // [S, 1024] bf16
    const __bf16* __restrict__ Kb,          // [S, 256]  bf16
    const __bf16* __restrict__ Vt,          // [256, S]  bf16 (transposed V)
    const unsigned char* __restrict__ mask, // [S] nonzero = masked out
    __bf16* __restrict__ attnb) {           // [S, 1024] bf16
  __shared__ __align__(16) __bf16 plds[8][16 * P_STRIDE];

  const int wave = threadIdx.x >> 5;
  const int lane = threadIdx.x & 31;
  const int hi = lane >> 4, n16 = lane & 15;
  const int h = blockIdx.y;
  const int hk = h >> 2;  // kv head = h / G
  const int s0 = blockIdx.x * 128 + wave * 16;

  const v16bf qa0 = load_frag_a(Qb, D_DIM, s0, h * DK_DIM);
  const v16bf qa1 = load_frag_a(Qb, D_DIM, s0, h * DK_DIM + 32);

  v8f o0 = v8f{}, o1 = v8f{}, o2 = v8f{}, o3 = v8f{};
  float mrow[8], lrow[8];
#pragma unroll
  for (int v = 0; v < 8; ++v) { mrow[v] = NEG_BIG; lrow[v] = 0.0f; }

  __bf16* myp = &plds[wave][0];

  // preload K fragments for tile 0
  v16bf kb0a = load_frag_b(Kb, DKV_DIM, 0, hk * DK_DIM);
  v16bf kb0b = load_frag_b(Kb, DKV_DIM, 0, hk * DK_DIM + 32);
  v16bf kb1a = load_frag_b(Kb, DKV_DIM, 16, hk * DK_DIM);
  v16bf kb1b = load_frag_b(Kb, DKV_DIM, 16, hk * DK_DIM + 32);

  for (int t0 = 0; t0 < S_LEN; t0 += 32) {
    // issue this tile's V fragments early (consumed after the softmax)
    v16bf vb0 = load_frag_b(Vt, S_LEN, hk * DK_DIM + 0, t0);
    v16bf vb1 = load_frag_b(Vt, S_LEN, hk * DK_DIM + 16, t0);
    v16bf vb2 = load_frag_b(Vt, S_LEN, hk * DK_DIM + 32, t0);
    v16bf vb3 = load_frag_b(Vt, S_LEN, hk * DK_DIM + 48, t0);

    // scores S[16x32] = Q[16x64] * K^T[64x32], two 16-col tiles
    v8f sA = v8f{};
    sA = wmma_bf16(qa0, kb0a, sA);
    sA = wmma_bf16(qa1, kb0b, sA);
    v8f sB = v8f{};
    sB = wmma_bf16(qa0, kb1a, sB);
    sB = wmma_bf16(qa1, kb1b, sB);

    // issue next tile's K fragments; softmax VALU work hides the latency
    if (t0 + 32 < S_LEN) {
      kb0a = load_frag_b(Kb, DKV_DIM, t0 + 32, hk * DK_DIM);
      kb0b = load_frag_b(Kb, DKV_DIM, t0 + 32, hk * DK_DIM + 32);
      kb1a = load_frag_b(Kb, DKV_DIM, t0 + 48, hk * DK_DIM);
      kb1b = load_frag_b(Kb, DKV_DIM, t0 + 48, hk * DK_DIM + 32);
    }

    const bool msk0 = mask[t0 + n16] != 0;
    const bool msk1 = mask[t0 + 16 + n16] != 0;

    // online softmax; rows live in VGPR index, cols across 16 lanes
#pragma unroll
    for (int v = 0; v < 8; ++v) {
      const float a = msk0 ? NEG_BIG : sA[v] * ATTN_SCALE;
      const float b = msk1 ? NEG_BIG : sB[v] * ATTN_SCALE;

      float rm = fmaxf(a, b);
#pragma unroll
      for (int off = 1; off < 16; off <<= 1)
        rm = fmaxf(rm, __shfl_xor(rm, off, 32));

      const float nm = fmaxf(mrow[v], rm);
      const float alpha = __expf(mrow[v] - nm);
      mrow[v] = nm;

      const float p0 = __expf(a - nm);
      const float p1 = __expf(b - nm);
      float rs = p0 + p1;
#pragma unroll
      for (int off = 1; off < 16; off <<= 1)
        rs += __shfl_xor(rs, off, 32);
      lrow[v] = lrow[v] * alpha + rs;

      o0[v] *= alpha; o1[v] *= alpha; o2[v] *= alpha; o3[v] *= alpha;

      const int r = v + 8 * hi;  // stage P (C-layout) for A-layout reload
      myp[r * P_STRIDE + n16] = f2bf(p0);
      myp[r * P_STRIDE + 16 + n16] = f2bf(p1);
    }

    asm volatile("s_wait_dscnt 0" ::: "memory");  // wave-private LDS transpose

    union { v16bf v; uv4 u[2]; } pf;
    const __bf16* pp = myp + n16 * P_STRIDE + hi * 16;
    pf.u[0] = *(const uv4*)(pp);
    pf.u[1] = *(const uv4*)(pp + 8);

    // O[16x64] += P[16x32] * V[32x64]  (V^T rows contiguous in t)
    o0 = wmma_bf16(pf.v, vb0, o0);
    o1 = wmma_bf16(pf.v, vb1, o1);
    o2 = wmma_bf16(pf.v, vb2, o2);
    o3 = wmma_bf16(pf.v, vb3, o3);
  }

#pragma unroll
  for (int v = 0; v < 8; ++v) {
    const float inv = 1.0f / lrow[v];
    const size_t r = (size_t)(s0 + v + 8 * hi) * D_DIM + h * DK_DIM + n16;
    attnb[r + 0]  = f2bf(o0[v] * inv);
    attnb[r + 16] = f2bf(o1[v] * inv);
    attnb[r + 32] = f2bf(o2[v] * inv);
    attnb[r + 48] = f2bf(o3[v] * inv);
  }
}

// ------------------------------ launch -------------------------------------

extern "C" void kernel_launch(void* const* d_in, const int* in_sizes, int n_in,
                              void* d_out, int out_size, void* d_ws, size_t ws_size,
                              hipStream_t stream) {
  (void)in_sizes; (void)n_in; (void)out_size; (void)ws_size;

  const float* x    = (const float*)d_in[0];                 // [S, D]
  const unsigned char* mask = (const unsigned char*)d_in[1]; // [S] bool
  const float* Wq = (const float*)d_in[2];
  const float* bq = (const float*)d_in[3];
  const float* Wk = (const float*)d_in[4];
  const float* bk = (const float*)d_in[5];
  const float* Wv = (const float*)d_in[6];
  const float* bv = (const float*)d_in[7];
  const float* Wo = (const float*)d_in[8];
  const float* bo = (const float*)d_in[9];
  float* out = (float*)d_out;

  char* ws = (char*)d_ws;
  size_t off = 0;
  auto alloc = [&](size_t bytes) {
    char* p = ws + off;
    off += (bytes + 255) & ~size_t(255);
    return p;
  };
  __bf16* xb    = (__bf16*)alloc((size_t)S_LEN * D_DIM * 2);
  __bf16* WqT   = (__bf16*)alloc((size_t)D_DIM * D_DIM * 2);
  __bf16* WkT   = (__bf16*)alloc((size_t)DKV_DIM * D_DIM * 2);
  __bf16* WvT   = (__bf16*)alloc((size_t)DKV_DIM * D_DIM * 2);
  __bf16* WoT   = (__bf16*)alloc((size_t)D_DIM * D_DIM * 2);
  __bf16* Qb    = (__bf16*)alloc((size_t)S_LEN * D_DIM * 2);
  __bf16* Kb    = (__bf16*)alloc((size_t)S_LEN * DKV_DIM * 2);
  __bf16* Vt    = (__bf16*)alloc((size_t)DKV_DIM * S_LEN * 2);
  __bf16* attnb = (__bf16*)alloc((size_t)S_LEN * D_DIM * 2);

  convert_f32_to_bf16_kernel<<<4096, 256, 0, stream>>>(x, xb, S_LEN * D_DIM);
  {
    dim3 blk(32, 8);
    transpose_to_bf16_kernel<<<dim3(D_DIM / 32, D_DIM / 32), blk, 0, stream>>>(
        Wq, WqT, D_DIM, D_DIM);
    transpose_to_bf16_kernel<<<dim3(DKV_DIM / 32, D_DIM / 32), blk, 0, stream>>>(
        Wk, WkT, D_DIM, DKV_DIM);
    transpose_to_bf16_kernel<<<dim3(DKV_DIM / 32, D_DIM / 32), blk, 0, stream>>>(
        Wv, WvT, D_DIM, DKV_DIM);
    transpose_to_bf16_kernel<<<dim3(D_DIM / 32, D_DIM / 32), blk, 0, stream>>>(
        Wo, WoT, D_DIM, D_DIM);
  }

  gemm_bf16_kernel<<<dim3(D_DIM / 64, S_LEN / 128), 256, 0, stream>>>(
      xb, WqT, bq, S_LEN, D_DIM, D_DIM, Qb, 0);
  gemm_bf16_kernel<<<dim3(DKV_DIM / 64, S_LEN / 128), 256, 0, stream>>>(
      xb, WkT, bk, S_LEN, DKV_DIM, D_DIM, Kb, 0);
  gemm_bf16_kernel<<<dim3(DKV_DIM / 64, S_LEN / 128), 256, 0, stream>>>(
      xb, WvT, bv, S_LEN, DKV_DIM, D_DIM, Vt, 1);  // stored transposed

  gqa_attn_kernel<<<dim3(S_LEN / 128, H_NUM), 256, 0, stream>>>(
      Qb, Kb, Vt, mask, attnb);

  gemm_bf16_kernel<<<dim3(D_DIM / 64, S_LEN / 128), 256, 0, stream>>>(
      attnb, WoT, bo, S_LEN, D_DIM, D_DIM, out, 2);
}
